// UserEncoder_7301444403690
// MI455X (gfx1250) — matrix operand order repaired and checked by reading, not compiled, not run on hardware
//
#include <hip/hip_runtime.h>
#include <hip/hip_bf16.h>
#include <math.h>

typedef __attribute__((ext_vector_type(16))) __bf16 v16bf;
typedef __attribute__((ext_vector_type(4)))  __bf16 v4bf;
typedef __attribute__((ext_vector_type(8)))  float  v8f;

namespace {
constexpr int BS   = 512;  // batches
constexpr int Lh   = 100;  // history length
constexpr int LH   = 128;  // history rows padded to multiple of 32 (K-chunks)
constexpr int LPad = 112;  // L padded to multiple of 16 (N tiles)
constexpr int NC   = 64;   // candidates
constexpr int Dm   = 256;  // news dim
constexpr int ICc  = 256;  // interest dim
constexpr int KC   = 32;   // max interests

// LDS layout (byte offsets). projT is dead after phase 2, so interests / w /
// masked-w alias its region; scores dead after phase 3. Total 145,408 B
// -> two workgroups co-resident per 320 KB WGP (4 waves/SIMD).
constexpr int OFF_H  = 0;                       // bf16 [LH][Dm]     65536
constexpr int OFF_PT = OFF_H  + LH * Dm * 2;    // bf16 [ICc][LPad]  57344
constexpr int OFF_IT = OFF_PT;                  // bf16 [KC][Dm]     16384 (aliases pT)
constexpr int OFF_W  = OFF_IT + KC * Dm * 2;    // f32  [NC][KC]      8192 (aliases pT)
constexpr int OFF_WM = OFF_W  + NC * KC * 4;    // bf16 [NC][KC]      4096 (aliases pT)
constexpr int OFF_SC = OFF_PT + ICc * LPad * 2; // f32  [KC][LPad]   14336
constexpr int OFF_AT = OFF_SC + KC * LPad * 4;  // bf16 [KC][LH]      8192
constexpr int SMEM_BYTES = OFF_AT + KC * LH * 2; // 145408 B
} // namespace

// native RNE f32 -> bf16 (emits v_cvt_pk_bf16_f32 on gfx1250)
__device__ __forceinline__ __bf16 f2bf(float f) { return (__bf16)f; }

// gfx1250 has a native V_TANH_F32 transcendental; prefer it over the ~30-op
// OCML expansion. Guarded so either toolchain still compiles.
__device__ __forceinline__ float fast_tanh(float x) {
#if __has_builtin(__builtin_amdgcn_tanhf)
  return __builtin_amdgcn_tanhf(x);
#elif __has_builtin(__builtin_amdgcn_tanh_f32)
  return __builtin_amdgcn_tanh_f32(x);
#else
  return tanhf(x);
#endif
}

__device__ __forceinline__ v8f wmma_bf16(v16bf a, v16bf b, v8f c) {
  // (neg_a, A, neg_b, B, c_mod, C, reuse_a, reuse_b)
  return __builtin_amdgcn_wmma_f32_16x16x32_bf16(false, a, false, b,
                                                 (short)0, c, false, false);
}

__global__ __launch_bounds__(256, 1) void
user_encoder_kernel(const float* __restrict__ hist,   // [BS][Lh][Dm]
                    const float* __restrict__ cand,   // [BS][NC][Dm]
                    const int*   __restrict__ nInterests, // [BS]
                    const int*   __restrict__ catCounts,  // [BS]
                    const float* __restrict__ Wg,     // [ICc][Dm]
                    const float* __restrict__ codes,  // [KC][ICc]
                    float* __restrict__ out)          // [BS][NC][Dm]
{
  extern __shared__ char smem[];
  __bf16* sh_h  = (__bf16*)(smem + OFF_H);
  __bf16* sh_pt = (__bf16*)(smem + OFF_PT);
  __bf16* sh_it = (__bf16*)(smem + OFF_IT);
  float*  sh_w  = (float*)(smem + OFF_W);
  __bf16* sh_wm = (__bf16*)(smem + OFF_WM);
  float*  sh_sc = (float*)(smem + OFF_SC);
  __bf16* sh_at = (__bf16*)(smem + OFF_AT);

  const int b    = blockIdx.x;
  const int tid  = threadIdx.x;
  const int wv   = tid >> 5;   // wave id, 0..7
  const int lane = tid & 31;
  const int lo   = lane & 15;  // lane within half
  const int hf   = lane >> 4;  // lane half (wave32 WMMA layout split)

  v8f vzero = {};

  // ---- Prefetch this batch's candidate block (64 KB) toward the caches so
  //      phase-5 A-fragment reads hit (global_prefetch_b8). ----
  {
    const char* cbase = (const char*)(cand + (size_t)b * NC * Dm);
    #pragma unroll
    for (int i = 0; i < (NC * Dm * 4) / (64 * 256); ++i)   // 4 lines/thread
      __builtin_prefetch(cbase + (i * 256 + tid) * 64, 0, 0);
  }

  // ---- Phase 0: history[b] f32 -> LDS bf16 [LH][Dm], rows >= Lh zeroed.
  //      float4 (B128) global loads, packed 4x bf16 (8B) LDS stores. ----
  {
    const float4* hsrc = (const float4*)(hist + (size_t)b * Lh * Dm);
    for (int idx = tid; idx < (LH * Dm) / 4; idx += 256) {
      int r = idx >> 6;                      // Dm/4 = 64 float4 per row
      float4 v;
      if (r < Lh) v = hsrc[idx];
      else        v = float4{0.0f, 0.0f, 0.0f, 0.0f};
      v4bf o;
      o[0] = f2bf(v.x); o[1] = f2bf(v.y); o[2] = f2bf(v.z); o[3] = f2bf(v.w);
      *(v4bf*)(sh_h + idx * 4) = o;
    }
  }
  __syncthreads();

  // ---- Phase 1: projT[c][l] = tanh(sum_d W[c,d] * H[l,d]) ----
  // 16 M-tiles (c) x 7 N-tiles (l); each wave owns 2 adjacent M-tiles and
  // processes them in one k-loop so every B fragment feeds 2 WMMAs.
  {
    const int c0 = wv * 32;                  // rows [c0, c0+32)
    v8f acc0[7], acc1[7];
    #pragma unroll
    for (int j = 0; j < 7; ++j) { acc0[j] = vzero; acc1[j] = vzero; }
    for (int k0 = 0; k0 < Dm; k0 += 32) {
      v16bf a0, a1;
      const float* wrow0 = Wg + (size_t)(c0 + lo) * Dm + k0;
      const float* wrow1 = wrow0 + 16 * Dm;
      #pragma unroll
      for (int i = 0; i < 16; ++i) {         // A: K contiguous 16 per lane half
        int k = 16 * hf + i;
        a0[i] = f2bf(wrow0[k]);
        a1[i] = f2bf(wrow1[k]);
      }
      #pragma unroll
      for (int j = 0; j < 7; ++j) {
        v16bf bb;                            // B[k][n] = H[l0+n][k0+k]
        const __bf16* hrow = sh_h + (size_t)(j * 16 + lo) * Dm + k0 + 16 * hf;
        #pragma unroll
        for (int i = 0; i < 16; ++i) bb[i] = hrow[i];
        acc0[j] = wmma_bf16(a0, bb, acc0[j]);
        acc1[j] = wmma_bf16(a1, bb, acc1[j]);
      }
    }
    #pragma unroll
    for (int j = 0; j < 7; ++j) {
      #pragma unroll
      for (int i = 0; i < 8; ++i) {
        int m = c0 + i + 8 * hf;
        int n = j * 16 + lo;
        sh_pt[m * LPad + n]        = f2bf(fast_tanh(acc0[j][i]));
        sh_pt[(m + 16) * LPad + n] = f2bf(fast_tanh(acc1[j][i]));
      }
    }
  }
  __syncthreads();

  // ---- Phase 2: scores[kc][l] = sum_c codes[kc,c] * projT[c,l] ----
  for (int tix = wv; tix < 14; tix += 8) {
    const int kc0 = (tix / 7) * 16;
    const int l0  = (tix % 7) * 16;
    v8f acc = vzero;
    for (int k0 = 0; k0 < ICc; k0 += 32) {
      v16bf a;
      const float* crow = codes + (size_t)(kc0 + lo) * ICc + k0;
      #pragma unroll
      for (int i = 0; i < 16; ++i) {
        int k = 16 * hf + i;
        a[i] = f2bf(crow[k]);
      }
      v16bf bb;
      #pragma unroll
      for (int i = 0; i < 16; ++i)
        bb[i] = sh_pt[(k0 + 16 * hf + i) * LPad + l0 + lo];
      acc = wmma_bf16(a, bb, acc);
    }
    #pragma unroll
    for (int i = 0; i < 8; ++i)
      sh_sc[(kc0 + i + 8 * hf) * LPad + l0 + lo] = acc[i];
  }
  __syncthreads();

  // ---- Phase 3: masked softmax over L. Rows kc >= nInt -> uniform 1/Lh ----
  {
    const int nInt = nInterests[b];
    if (tid < KC) {
      const int r = tid;
      float*  srow = sh_sc + r * LPad;
      __bf16* arow = sh_at + r * LH;
      if (r >= nInt) {
        const float u = 1.0f / (float)Lh;
        for (int l = 0; l < LH; ++l) arow[l] = f2bf((l < Lh) ? u : 0.0f);
      } else {
        float mx = -3.0e38f;
        for (int l = 0; l < Lh; ++l) mx = fmaxf(mx, srow[l]);
        float s = 0.0f;
        for (int l = 0; l < Lh; ++l) s += __expf(srow[l] - mx);
        float inv = 1.0f / s;
        for (int l = 0; l < LH; ++l)
          arow[l] = f2bf((l < Lh) ? __expf(srow[l] - mx) * inv : 0.0f);
      }
    }
  }
  __syncthreads();   // also covers WAR: phase 4 overwrites pT-aliased sh_it

  // ---- Phase 4: interests[kc][d] = sum_l attn[kc,l] * H[l,d] ----
  for (int tix = wv; tix < 32; tix += 8) {
    const int kc0 = (tix & 1) * 16;
    const int d0  = (tix >> 1) * 16;
    v8f acc = vzero;
    for (int k0 = 0; k0 < LH; k0 += 32) {
      v16bf a;
      const __bf16* arow = sh_at + (kc0 + lo) * LH + k0;
      #pragma unroll
      for (int i = 0; i < 16; ++i) {
        int k = 16 * hf + i;
        a[i] = arow[k];
      }
      v16bf bb;
      #pragma unroll
      for (int i = 0; i < 16; ++i)
        bb[i] = sh_h[(size_t)(k0 + 16 * hf + i) * Dm + d0 + lo];
      acc = wmma_bf16(a, bb, acc);
    }
    #pragma unroll
    for (int i = 0; i < 8; ++i)
      sh_it[(kc0 + i + 8 * hf) * Dm + d0 + lo] = f2bf(acc[i]);
  }
  __syncthreads();

  // ---- Phase 5: w[n][kc] = sum_d cand[n,d] * interests[kc,d] (1 tile/wave) --
  {
    const int n0  = (wv >> 1) * 16;
    const int kc0 = (wv & 1) * 16;
    v8f acc = vzero;
    for (int k0 = 0; k0 < Dm; k0 += 32) {
      v16bf a;
      const float* crow = cand + ((size_t)b * NC + n0 + lo) * Dm + k0;
      #pragma unroll
      for (int i = 0; i < 16; ++i) {
        int k = 16 * hf + i;
        a[i] = f2bf(crow[k]);
      }
      v16bf bb;                  // B[k][n] = interests[kc0+n][k0+k]
      const __bf16* irow = sh_it + (kc0 + lo) * Dm + k0 + 16 * hf;
      #pragma unroll
      for (int i = 0; i < 16; ++i) bb[i] = irow[i];
      acc = wmma_bf16(a, bb, acc);
    }
    #pragma unroll
    for (int i = 0; i < 8; ++i)
      sh_w[(n0 + i + 8 * hf) * KC + kc0 + lo] = acc[i];
  }
  __syncthreads();

  // ---- Phase 6: per-candidate variable top-k mask over KC values ----
  if (tid < NC) {
    const int r = tid;
    float cc = (float)catCounts[b];
    int dynK = (int)ceilf(log2f(8.0f * cc));   // M_PARAM = 8
    dynK = dynK < 1 ? 1 : (dynK > KC ? KC : dynK);
    float tmp[KC];
    #pragma unroll
    for (int j = 0; j < KC; ++j) tmp[j] = sh_w[r * KC + j];
    unsigned keep = 0u;
    for (int it = 0; it < dynK; ++it) {
      int am = 0; float mv = -3.4e38f;
      #pragma unroll
      for (int j = 0; j < KC; ++j) {
        bool taken = (keep >> j) & 1u;
        if (!taken && tmp[j] > mv) { mv = tmp[j]; am = j; }
      }
      keep |= (1u << am);
    }
    #pragma unroll
    for (int j = 0; j < KC; ++j)
      sh_wm[r * KC + j] = ((keep >> j) & 1u) ? f2bf(tmp[j]) : f2bf(0.0f);
  }
  __syncthreads();

  // ---- Phase 7: user[n][d] = sum_kc masked[n,kc] * interests[kc,d] ----
  // K = 32 is exactly one WMMA chunk; 64 tiles, 8/wave; write straight to HBM.
  for (int tix = wv; tix < 64; tix += 8) {
    const int n0 = (tix & 3) * 16;
    const int d0 = (tix >> 2) * 16;
    v16bf a;
    const __bf16* wrow = sh_wm + (n0 + lo) * KC;
    #pragma unroll
    for (int i = 0; i < 16; ++i) {
      int k = 16 * hf + i;
      a[i] = wrow[k];
    }
    v16bf bb;
    #pragma unroll
    for (int i = 0; i < 16; ++i)
      bb[i] = sh_it[(16 * hf + i) * Dm + d0 + lo];
    v8f acc = vzero;
    acc = wmma_bf16(a, bb, acc);
    #pragma unroll
    for (int i = 0; i < 8; ++i)
      out[((size_t)b * NC + n0 + i + 8 * hf) * Dm + d0 + lo] = acc[i];
  }
}

extern "C" void kernel_launch(void* const* d_in, const int* in_sizes, int n_in,
                              void* d_out, int out_size, void* d_ws, size_t ws_size,
                              hipStream_t stream) {
  (void)in_sizes; (void)n_in; (void)out_size; (void)d_ws; (void)ws_size;
  const float* hist  = (const float*)d_in[0];
  // d_in[1] = history_mask (unused by the reference forward)
  const float* cand  = (const float*)d_in[2];
  const int*   nint  = (const int*)d_in[3];
  const int*   ccnt  = (const int*)d_in[4];
  const float* Wg    = (const float*)d_in[5];
  const float* codes = (const float*)d_in[6];
  float* out = (float*)d_out;

  user_encoder_kernel<<<dim3(BS), dim3(256), SMEM_BYTES, stream>>>(
      hist, cand, nint, ccnt, Wg, codes, out);
}
